// LigerFusedLinearPPOLoss_31568009625762
// MI455X (gfx1250) — matrix operand order repaired and checked by reading, not compiled, not run on hardware
//
#include <hip/hip_runtime.h>

// ---------------------------------------------------------------------------
// Problem constants (from reference): B=4, T=512, H=2048, V=32000
// ---------------------------------------------------------------------------
#define HDIM   2048
#define TOK    2048          // B*T
#define VOC    32000
#define NCHUNK 25
#define CHUNK  1280          // VOC / NCHUNK
#define KSTEPS 64            // HDIM / 32
#define LDSTR  2056          // padded bf16 row stride (16B-aligned, bank-friendly)
#define IGNORE_INDEX (-100)

typedef __attribute__((ext_vector_type(16))) __bf16        v16bf;
typedef __attribute__((ext_vector_type(8)))  float         v8f;
typedef __attribute__((ext_vector_type(4)))  unsigned int  u32x4;
typedef __attribute__((ext_vector_type(4)))  float         f32x4;

union FragA { v16bf bf; u32x4 q[2]; };
union FragB { v16bf bf; unsigned int u[8]; };

// round-to-nearest-even f32 -> bf16 (used once for the A staging pass)
__device__ inline unsigned short f2bf_rne(float f) {
  unsigned u = __float_as_uint(f);
  unsigned r = u + 0x7FFFu + ((u >> 16) & 1u);
  return (unsigned short)(r >> 16);
}

// Pack two f32 high-halves into one dword (truncating bf16) with one v_perm_b32.
__device__ inline unsigned pack2_bf16(float lo, float hi) {
  return __builtin_amdgcn_perm(__float_as_uint(hi), __float_as_uint(lo), 0x07060302u);
}

// Load 16 consecutive f32 of a weight row, convert to a bf16 B-fragment.
// B layout (32x16, bf16): lanes 0-15 hold K = k0..k0+15 for N=lane,
// lanes 16-31 hold K = k0+16..k0+31 for N=lane-16 (caller picks base).
__device__ inline v16bf load_bfrag(const float* __restrict__ p) {
  const f32x4* p4 = (const f32x4*)p;
  FragB fb;
#pragma unroll
  for (int i = 0; i < 4; ++i) {
    f32x4 v = p4[i];
    fb.u[i * 2 + 0] = pack2_bf16(v[0], v[1]);
    fb.u[i * 2 + 1] = pack2_bf16(v[2], v[3]);
  }
  return fb.bf;
}

// ---------------------------------------------------------------------------
// Kernel 1: fused GEMM + online row logsumexp + target-logit gather.
// grid = (128 M-tiles, 25 vocab chunks, 2 models), block = 256 (8 waves).
// Writes (rowmax, rowsumexp, targetlogit) per (model, row, chunk) to ws.
// ---------------------------------------------------------------------------
__global__ __launch_bounds__(256) void ppo_logits_partials(
    const float* __restrict__ x,  const float* __restrict__ W,
    const float* __restrict__ bv, const int*   __restrict__ target,
    const float* __restrict__ xr, const float* __restrict__ Wr,
    const float* __restrict__ bvr, float* __restrict__ part) {
  const int mtile = blockIdx.x;   // 0..127
  const int chunk = blockIdx.y;   // 0..24
  const int model = blockIdx.z;   // 0..1
  const float* __restrict__ X  = model ? xr  : x;
  const float* __restrict__ Wm = model ? Wr  : W;
  const float* __restrict__ Bv = model ? bvr : bv;

  __shared__ unsigned short ldsA[16 * LDSTR];   // 16 x 2048 bf16 A-tile (padded)
  __shared__ float comb[8][16][3];              // per-wave row partials

  const int tid = threadIdx.x;
  const int m0  = mtile * 16;

  // ---- stage activation tile to LDS as bf16 (coalesced f32 reads) ----
  for (int i = tid; i < 16 * HDIM; i += 256) {
    int r = i >> 11, c = i & (HDIM - 1);
    ldsA[r * LDSTR + c] = f2bf_rne(X[(size_t)(m0 + r) * HDIM + c]);
  }
  __syncthreads();

  const int lane = tid & 31;
  const int wave = tid >> 5;
  const int half = lane >> 4;    // K-half / M-half selector
  const int l16  = lane & 15;

  // C/D layout: lanes 0-15 -> rows m0+r, lanes 16-31 -> rows m0+8+r (r = vgpr)
  float rmax[8], rsum[8], rtgt[8];
  int   tgts[8];
#pragma unroll
  for (int r = 0; r < 8; ++r) {
    rmax[r] = -INFINITY; rsum[r] = 0.f; rtgt[r] = -INFINITY;
    tgts[r] = target[m0 + half * 8 + r];
  }

  const unsigned short* aRow = ldsA + l16 * LDSTR;   // A row for this lane
  const int aoff = half * 16;                        // A K-subblock offset

  // 80 N-tiles per chunk -> each wave: 5 iterations x 2 tiles
  for (int it = 0; it < 5; ++it) {
    const int t0 = wave * 2 + it * 16;               // tile pair index
    const int n0 = chunk * CHUNK + t0 * 16;          // global vocab base
    const float* __restrict__ wrow0 = Wm + (size_t)(n0 + l16) * HDIM;
    const float* __restrict__ wrow1 = Wm + (size_t)(n0 + 16 + l16) * HDIM;

    v8f acc0 = {}; v8f acc1 = {};
#pragma unroll 4
    for (int ks = 0; ks < KSTEPS; ++ks) {
      const int k0 = ks * 32;
      // A fragment: lanes 0-15 take K = k0+0..15; lanes 16-31 take K = k0+16..31
      FragA a;
      const u32x4* pa = (const u32x4*)(aRow + k0 + aoff);
      a.q[0] = pa[0];          // 8 bf16
      a.q[1] = pa[1];          // +8 elements = +16B
      // B fragments: 16 consecutive f32 of the lane's weight row
      const int kb = k0 + half * 16;
      v16bf b0 = load_bfrag(wrow0 + kb);
      v16bf b1 = load_bfrag(wrow1 + kb);
      acc0 = __builtin_amdgcn_wmma_f32_16x16x32_bf16(false, a.bf, false, b0,
                                                     (short)0, acc0, false, false);
      acc1 = __builtin_amdgcn_wmma_f32_16x16x32_bf16(false, a.bf, false, b1,
                                                     (short)0, acc1, false, false);
    }

    // fold this tile pair into the online (max, sumexp) state + target gather
    const float b0v = Bv[n0 + l16];
    const float b1v = Bv[n0 + 16 + l16];
    const int col0 = n0 + l16, col1 = n0 + 16 + l16;
#pragma unroll
    for (int r = 0; r < 8; ++r) {
      float v0 = acc0[r] + b0v;
      float v1 = acc1[r] + b1v;
      float nm = fmaxf(rmax[r], fmaxf(v0, v1));
      rsum[r] = rsum[r] * __expf(rmax[r] - nm) + __expf(v0 - nm) + __expf(v1 - nm);
      rmax[r] = nm;
      if (tgts[r] == col0) rtgt[r] = v0;
      if (tgts[r] == col1) rtgt[r] = v1;
    }
  }

  // ---- butterfly reduce across the 16 lanes of each half (N direction) ----
#pragma unroll
  for (int mask = 1; mask <= 8; mask <<= 1) {
#pragma unroll
    for (int r = 0; r < 8; ++r) {
      float om = __shfl_xor(rmax[r], mask, 32);
      float os = __shfl_xor(rsum[r], mask, 32);
      float ot = __shfl_xor(rtgt[r], mask, 32);
      float nm = fmaxf(rmax[r], om);
      rsum[r] = rsum[r] * __expf(rmax[r] - nm) + os * __expf(om - nm);
      rmax[r] = nm;
      rtgt[r] = fmaxf(rtgt[r], ot);
    }
  }

  if (l16 == 0) {
#pragma unroll
    for (int r = 0; r < 8; ++r) {
      comb[wave][half * 8 + r][0] = rmax[r];
      comb[wave][half * 8 + r][1] = rsum[r];
      comb[wave][half * 8 + r][2] = rtgt[r];
    }
  }
  __syncthreads();

  // ---- combine the 8 waves, one thread per row, write chunk partial ----
  if (tid < 16) {
    float gm = comb[0][tid][0], gs = comb[0][tid][1], gt = comb[0][tid][2];
#pragma unroll
    for (int w = 1; w < 8; ++w) {
      float om = comb[w][tid][0], os = comb[w][tid][1], ot = comb[w][tid][2];
      float nm = fmaxf(gm, om);
      gs = gs * __expf(gm - nm) + os * __expf(om - nm);
      gm = nm;
      gt = fmaxf(gt, ot);
    }
    size_t idx = (((size_t)model * TOK + (m0 + tid)) * NCHUNK + chunk) * 3;
    part[idx + 0] = gm;
    part[idx + 1] = gs;
    part[idx + 2] = gt;
  }
}

// ---------------------------------------------------------------------------
// Kernel 2: merge chunk partials -> per-token logp for both models,
// PPO clip objective, deterministic tree reduction to the scalar mean.
// ---------------------------------------------------------------------------
__global__ __launch_bounds__(1024) void ppo_loss_reduce(
    const float* __restrict__ part, const int* __restrict__ target,
    const unsigned char* __restrict__ pref, float* __restrict__ out) {
  __shared__ float red[1024];
  const int tid = threadIdx.x;
  float lsum = 0.f;

  for (int tok = tid; tok < TOK; tok += 1024) {
    float lp[2];
#pragma unroll
    for (int mdl = 0; mdl < 2; ++mdl) {
      size_t base = ((size_t)mdl * TOK + tok) * NCHUNK * 3;
      float gm = part[base + 0], gs = part[base + 1], gt = part[base + 2];
      for (int c = 1; c < NCHUNK; ++c) {
        float om = part[base + c * 3 + 0];
        float os = part[base + c * 3 + 1];
        float ot = part[base + c * 3 + 2];
        float nm = fmaxf(gm, om);
        gs = gs * __expf(gm - nm) + os * __expf(om - nm);
        gm = nm;
        gt = fmaxf(gt, ot);
      }
      float lse = gm + __logf(gs);
      lp[mdl] = (target[tok] == IGNORE_INDEX) ? 0.f : (gt - lse);
    }
    const int b = tok >> 9;                      // tok / T, T=512
    const float adv = pref[b] ? 1.0f : -0.01f;
    const float ratio = __expf(lp[0] - lp[1]);
    const float unc = ratio * adv;
    const float cl  = fminf(fmaxf(ratio, 0.8f), 1.2f) * adv;
    lsum += -fminf(unc, cl);
  }

  red[tid] = lsum;
  __syncthreads();
  for (int s = 512; s > 0; s >>= 1) {
    if (tid < s) red[tid] += red[tid + s];
    __syncthreads();
  }
  if (tid == 0) out[0] = red[0] * (1.0f / (float)TOK);
}

// ---------------------------------------------------------------------------
extern "C" void kernel_launch(void* const* d_in, const int* in_sizes, int n_in,
                              void* d_out, int out_size, void* d_ws, size_t ws_size,
                              hipStream_t stream) {
  const float*         x    = (const float*)d_in[0];
  const float*         W    = (const float*)d_in[1];
  const float*         bv   = (const float*)d_in[2];
  const int*           tgt  = (const int*)d_in[3];
  const unsigned char* pref = (const unsigned char*)d_in[4];
  const float*         xr   = (const float*)d_in[5];
  const float*         Wr   = (const float*)d_in[6];
  const float*         bvr  = (const float*)d_in[7];
  float* part = (float*)d_ws;             // 2*2048*25*3 floats = 1.23 MB
  float* out  = (float*)d_out;

  dim3 grid(TOK / 16, NCHUNK, 2);
  ppo_logits_partials<<<grid, 256, 0, stream>>>(x, W, bv, tgt, xr, Wr, bvr, part);
  ppo_loss_reduce<<<1, 1024, 0, stream>>>(part, tgt, pref, out);
}
